// RnnMsgAgent_14843406975264
// MI455X (gfx1250) — matrix operand order repaired and checked by reading, not compiled, not run on hardware
//
#include <hip/hip_runtime.h>
#include <stdint.h>

#define BB   16384
#define NAG  8
#define NQv  64
#define HID  256
#define AA   30
#define TB   4          // batches per workgroup
#define ROWS 32         // TB*NAG
#define LDA  328        // sA stride in bf16 elems (320 + pad)
#define LDC  264        // sC stride in bf16 elems (256 + pad)
#define LDE  36         // sEv stride in f32 elems (32 + pad, mult of 4)

typedef __attribute__((ext_vector_type(16))) __bf16    v16bf;
typedef __attribute__((ext_vector_type(8)))  float     v8f;
typedef __attribute__((ext_vector_type(4)))  uint32_t  u32x4;

union AFrag { v16bf v; uint32_t u[8]; };
union BFrag { v16bf v; u32x4 q[2]; };

__device__ __forceinline__ uint32_t bf16b(float f) {
  uint32_t u = __float_as_uint(f);
  u = (u + 0x7FFFu + ((u >> 16) & 1u)) >> 16;   // round-to-nearest-even
  return u;
}
__device__ __forceinline__ uint32_t pack2(float lo, float hi) {
  return bf16b(lo) | (bf16b(hi) << 16);
}

// ---- Pre-swizzle W_comb [320][256] f32 -> bf16 WMMA B-fragment layout ----
// out[((kt*16+ct)*32 + lane)*8 + r] = pack( W[kb,col], W[kb+1,col] )
//   col = ct*16 + (lane&15),  kb = kt*32 + (lane>>4)*16 + 2r
__global__ void swizzle_w1(const float* __restrict__ W, uint32_t* __restrict__ out) {
  int idx  = blockIdx.x * 256 + threadIdx.x;       // 0..40959
  int r    = idx & 7;
  int lane = (idx >> 3) & 31;
  int tile = idx >> 8;                              // kt*16 + ct, 0..159
  int kt = tile >> 4, ct = tile & 15;
  int col = ct * 16 + (lane & 15);
  int kb  = kt * 32 + (lane >> 4) * 16 + 2 * r;
  out[idx] = pack2(W[kb * 256 + col], W[(kb + 1) * 256 + col]);
}

// ---- Pre-swizzle W2 [256][30] f32 -> bf16 B-fragments, N padded to 32 ----
__global__ void swizzle_w2(const float* __restrict__ W, uint32_t* __restrict__ out) {
  int idx  = blockIdx.x * 256 + threadIdx.x;       // 0..4095
  int r    = idx & 7;
  int lane = (idx >> 3) & 31;
  int tile = idx >> 8;                              // kt*2 + ct, 0..15
  int kt = tile >> 1, ct = tile & 1;
  int col = ct * 16 + (lane & 15);
  int kb  = kt * 32 + (lane >> 4) * 16 + 2 * r;
  float lo = (col < AA) ? W[kb * AA + col] : 0.f;
  float hi = (col < AA) ? W[(kb + 1) * AA + col] : 0.f;
  out[idx] = pack2(lo, hi);
}

__global__ __launch_bounds__(256)
void fused_agent_kernel(const float* __restrict__ q_g, const float* __restrict__ k_g,
                        const float* __restrict__ v_g, const float* __restrict__ h_g,
                        const int*   __restrict__ st_g, const float* __restrict__ wat,
                        const float* __restrict__ batt, const float* __restrict__ bcomb,
                        const float* __restrict__ b2v, const uint32_t* __restrict__ w1s,
                        const uint32_t* __restrict__ w2s, float* __restrict__ out)
{
  __shared__ __align__(16) uint16_t sA[ROWS * LDA];   // [applied | hidden] bf16
  __shared__ __align__(16) uint16_t sC[ROWS * LDC];   // combined (relu) bf16
  __shared__ __align__(16) float    sEv[ROWS * LDE];  // evidence f32 (cols 30,31 = 0)
  __shared__ float sLog[TB][NAG][NAG];
  __shared__ float sAttn[TB][NAG][NAG];
  __shared__ float sKb[TB][NAG];

  const int t   = threadIdx.x;
  const int gb0 = blockIdx.x * TB;

  // ---- hidden -> sA[:, 64:320] as bf16 (coalesced) ----
  for (int it = 0; it < 32; ++it) {
    int e = t + 256 * it;                 // 0..8191 over [b][i][c]
    int b = e >> 11, rem = e & 2047;
    int i = rem >> 8, c = rem & 255;
    float val = h_g[(size_t)gb0 * NAG * HID + e];
    sA[(b * NAG + i) * LDA + NQv + c] = (uint16_t)bf16b(val);
  }

  // ---- key bias: kb[b][j] = b_attn[j] + sum_k keys_flat[b][k]*W_attn[64+k][j] ----
  {
    int p = t >> 3, sub = t & 7;
    int b = p >> 3, j = p & 7;
    const float* kp = k_g + (size_t)(gb0 + b) * 512 + sub * 64;
    float acc = 0.f;
#pragma unroll
    for (int m = 0; m < 16; ++m) {
      float4 kv = *(const float4*)(kp + 4 * m);
      int kk = NQv + sub * 64 + 4 * m;
      acc += kv.x * wat[(kk + 0) * NAG + j];
      acc += kv.y * wat[(kk + 1) * NAG + j];
      acc += kv.z * wat[(kk + 2) * NAG + j];
      acc += kv.w * wat[(kk + 3) * NAG + j];
    }
    acc += __shfl_xor(acc, 1, 32);
    acc += __shfl_xor(acc, 2, 32);
    acc += __shfl_xor(acc, 4, 32);
    if (sub == 0) sKb[b][j] = acc + batt[j];
  }
  __syncthreads();

  // ---- logits[b][i][j] = q[b,i,:].W_attn[0:64,j] + kb[b][j] ----
  {
    int b = t >> 6, i = (t >> 3) & 7, j = t & 7;
    const float* qp = q_g + ((size_t)(gb0 + b) * NAG + i) * NQv;
    float acc = 0.f;
#pragma unroll
    for (int m = 0; m < 16; ++m) {
      float4 qv = *(const float4*)(qp + 4 * m);
      acc += qv.x * wat[(4 * m + 0) * NAG + j];
      acc += qv.y * wat[(4 * m + 1) * NAG + j];
      acc += qv.z * wat[(4 * m + 2) * NAG + j];
      acc += qv.w * wat[(4 * m + 3) * NAG + j];
    }
    sLog[b][i][j] = acc + sKb[b][j];
  }
  __syncthreads();

  // ---- softmax over j (8 wide) ----
  {
    int b = t >> 6, i = (t >> 3) & 7, j = t & 7;
    float l[8];
#pragma unroll
    for (int x = 0; x < 8; ++x) l[x] = sLog[b][i][x];
    float mx = l[0];
#pragma unroll
    for (int x = 1; x < 8; ++x) mx = fmaxf(mx, l[x]);
    float s = 0.f;
#pragma unroll
    for (int x = 0; x < 8; ++x) s += __expf(l[x] - mx);
    sAttn[b][i][j] = __expf(l[j] - mx) / s;
  }
  __syncthreads();

  // ---- applied = attn @ value -> sA[:, 0:64] bf16 ----
  {
    int b = t >> 6, i = (t >> 3) & 7, vb = t & 7;
    int v0 = vb * 8;
    float a8[8] = {0, 0, 0, 0, 0, 0, 0, 0};
#pragma unroll
    for (int j = 0; j < 8; ++j) {
      float a = sAttn[b][i][j];
      const float* vp = v_g + ((size_t)(gb0 + b) * NAG + j) * NQv + v0;
      float4 x = *(const float4*)vp;
      float4 y = *(const float4*)(vp + 4);
      a8[0] += a * x.x; a8[1] += a * x.y; a8[2] += a * x.z; a8[3] += a * x.w;
      a8[4] += a * y.x; a8[5] += a * y.y; a8[6] += a * y.z; a8[7] += a * y.w;
    }
    int row = b * NAG + i;
#pragma unroll
    for (int m = 0; m < 4; ++m)
      *(uint32_t*)&sA[row * LDA + v0 + 2 * m] = pack2(a8[2 * m], a8[2 * m + 1]);
  }
  __syncthreads();

  const int lane = t & 31, wid = t >> 5;
  const int lane16 = lane & 15, hi8 = (lane >> 4) * 8;
  const int koffs[8] = {0, 2, 4, 6, 8, 10, 12, 14};

  // ---- GEMM-1: combined = relu([applied|hidden](32x320) @ W_comb(320x256)) ----
  {
    const int rowT = wid >> 2, colT = wid & 3;
    const int m0 = rowT * 16;
    v8f zz = {0, 0, 0, 0, 0, 0, 0, 0};
    v8f acc[4]; acc[0] = zz; acc[1] = zz; acc[2] = zz; acc[3] = zz;
    for (int kt = 0; kt < 10; ++kt) {
      AFrag af;
      const uint16_t* ap = &sA[(m0 + lane16) * LDA + kt * 32 + 2 * hi8];
#pragma unroll
      for (int r = 0; r < 8; ++r) af.u[r] = *(const uint32_t*)(ap + koffs[r]);
#pragma unroll
      for (int c = 0; c < 4; ++c) {
        const uint32_t* bp = w1s + ((size_t)(kt * 16 + colT * 4 + c) * 32 + lane) * 8;
        BFrag bf;
        bf.q[0] = *(const u32x4*)bp;
        bf.q[1] = *(const u32x4*)(bp + 4);
        acc[c] = __builtin_amdgcn_wmma_f32_16x16x32_bf16(
            false, af.v, false, bf.v, (short)0, acc[c], false, false);
      }
    }
#pragma unroll
    for (int c = 0; c < 4; ++c) {
      int col = colT * 64 + c * 16 + lane16;
      float bias = bcomb[col];
#pragma unroll
      for (int r = 0; r < 8; ++r) {
        int row = m0 + hi8 + r;
        float v = fmaxf(acc[c][r] + bias, 0.f);
        sC[row * LDC + col] = (uint16_t)bf16b(v);
      }
    }
  }
  __syncthreads();

  // ---- GEMM-2: evidence = clip(combined(32x256) @ W2(256x30), 0) ----
  if (wid < 4) {
    const int rt = wid >> 1, ct2 = wid & 1;
    const int m0 = rt * 16, n0 = ct2 * 16;
    v8f acc = {0, 0, 0, 0, 0, 0, 0, 0};
    for (int kt = 0; kt < 8; ++kt) {
      AFrag af;
      const uint16_t* ap = &sC[(m0 + lane16) * LDC + kt * 32 + 2 * hi8];
#pragma unroll
      for (int r = 0; r < 8; ++r) af.u[r] = *(const uint32_t*)(ap + koffs[r]);
      const uint32_t* bp = w2s + ((size_t)(kt * 2 + ct2) * 32 + lane) * 8;
      BFrag bf;
      bf.q[0] = *(const u32x4*)bp;
      bf.q[1] = *(const u32x4*)(bp + 4);
      acc = __builtin_amdgcn_wmma_f32_16x16x32_bf16(
          false, af.v, false, bf.v, (short)0, acc, false, false);
    }
    int col = n0 + lane16;
    float bias = (col < AA) ? b2v[col] : 0.f;
#pragma unroll
    for (int r = 0; r < 8; ++r) {
      int row = m0 + hi8 + r;
      sEv[row * LDE + col] = fmaxf(acc[r] + bias, 0.f);   // cols 30,31 -> 0
    }
  }
  __syncthreads();

  // ---- Dempster-Shafer message combine: one (batch, receiver) per 8 lanes ----
  {
    const int base8 = lane & 24;
    const int p = t >> 3, sub = t & 7;
    const int b = p >> 3, j = p & 7;
    const int gb = gb0 + b;
    const int rowj = b * NAG + j;
    const int rowh = b * NAG + sub;

    int stv = st_g[((size_t)gb * NAG + j) * NAG + sub];
    float mk = (stv != 0 || sub == j) ? 1.f : 0.f;

    // S for agent h==sub (pad cols are zero)
    float ss = 0.f;
    const float4* evp = (const float4*)&sEv[rowh * LDE];
#pragma unroll
    for (int m = 0; m < 8; ++m) { float4 x = evp[m]; ss += x.x + x.y + x.z + x.w; }
    float S_own = ss + 30.f;

    float evj[4];
#pragma unroll
    for (int m = 0; m < 4; ++m) evj[m] = sEv[rowj * LDE + sub + 8 * m];

    float b0[4] = {0, 0, 0, 0};
    float u0 = 1.f, sb0 = 0.f;
    for (int h = 0; h < 8; ++h) {
      float Sh = __shfl(S_own, base8 + h, 32);
      float mh = __shfl(mk, base8 + h, 32);
      float invS = mh / Sh;                        // 0 if unmasked
      float u1 = mh * (30.f / Sh) + (1.f - mh);    // cand_u
      float b1[4], dotp = 0.f;
#pragma unroll
      for (int m = 0; m < 4; ++m) {
        b1[m] = sEv[(b * NAG + h) * LDE + sub + 8 * m] * invS;
        dotp += b0[m] * b1[m];
      }
      dotp += __shfl_xor(dotp, 1, 32);
      dotp += __shfl_xor(dotp, 2, 32);
      dotp += __shfl_xor(dotp, 4, 32);
      float sb1 = mh * (Sh - 30.f) / Sh;           // sum(cand_b) analytically
      float C   = sb0 * sb1 - dotp;
      float inv = 1.f / (1.f - C);
#pragma unroll
      for (int m = 0; m < 4; ++m)
        b0[m] = (b0[m] * b1[m] + b0[m] * u1 + b1[m] * u0) * inv;
      sb0 = (dotp + sb0 * u1 + sb1 * u0) * inv;
      u0  = u0 * u1 * inv;
    }
    float scale = 30.f / u0;                       // belief_to_evidence
    float s2p = 0.f;
    float ev2v[4];
#pragma unroll
    for (int m = 0; m < 4; ++m) {
      ev2v[m] = 2.f * evj[m] + 0.1f * (b0[m] * scale);  // ev + (ev + 0.1*comb)
      s2p += ev2v[m];
    }
    size_t obase = ((size_t)gb * NAG + j) * AA;
#pragma unroll
    for (int m = 0; m < 4; ++m) {
      int a = sub + 8 * m;
      if (a < AA) out[obase + a] = ev2v[m];
    }
    s2p += __shfl_xor(s2p, 1, 32);
    s2p += __shfl_xor(s2p, 2, 32);
    s2p += __shfl_xor(s2p, 4, 32);
    float S2 = s2p + 30.f;
    float Sj = __shfl(S_own, base8 + j, 32);
    if (sub == 0)
      out[(size_t)BB * NAG * AA + (size_t)gb * NAG + j] = 30.f / S2 - 30.f / Sj;
  }
}

extern "C" void kernel_launch(void* const* d_in, const int* in_sizes, int n_in,
                              void* d_out, int out_size, void* d_ws, size_t ws_size,
                              hipStream_t stream) {
  const float* q   = (const float*)d_in[0];
  const float* k   = (const float*)d_in[1];
  const float* v   = (const float*)d_in[2];
  const float* h   = (const float*)d_in[3];
  const int*   st  = (const int*)d_in[4];
  const float* wat = (const float*)d_in[5];
  const float* ba  = (const float*)d_in[6];
  const float* wc  = (const float*)d_in[7];
  const float* bc  = (const float*)d_in[8];
  const float* w2  = (const float*)d_in[9];
  const float* b2  = (const float*)d_in[10];

  uint32_t* w1s = (uint32_t*)d_ws;          // 40960 u32 = 160 KB
  uint32_t* w2s = w1s + 40960;              //  4096 u32 =  16 KB

  swizzle_w1<<<160, 256, 0, stream>>>(wc, w1s);
  swizzle_w2<<<16, 256, 0, stream>>>(w2, w2s);
  fused_agent_kernel<<<BB / TB, 256, 0, stream>>>(q, k, v, h, st, wat, ba, bc, b2,
                                                  w1s, w2s, (float*)d_out);
}
